// RoPETransformerEncoderLayer_3435973837500
// MI455X (gfx1250) — matrix operand order repaired and checked by reading, not compiled, not run on hardware
//
#include <hip/hip_runtime.h>
#include <math.h>

// ---------------------------------------------------------------------------
// RoPE Transformer encoder layer for MI455X (gfx1250, wave32, WMMA + TDM).
// GEMMs via v_wmma_f32_16x16x32_f16; tiles staged into LDS by the Tensor
// Data Mover (tensor_load_to_lds, TENSORcnt) with double buffering.
// ---------------------------------------------------------------------------

#define BB 2
#define SS 2048
#define HH 1024
#define NHH 16
#define HDD 64
#define FFF 4096

typedef __attribute__((ext_vector_type(16))) _Float16 v16h;
typedef __attribute__((ext_vector_type(8)))  float    v8f;
typedef __attribute__((ext_vector_type(4)))  unsigned int v4u;
typedef __attribute__((ext_vector_type(4)))  int v4i;
typedef __attribute__((ext_vector_type(8)))  int v8i;

#if defined(__has_builtin)
#  if __has_builtin(__builtin_amdgcn_tensor_load_to_lds) && \
      __has_builtin(__builtin_amdgcn_s_wait_tensorcnt)
#    define USE_TDM 1
#  endif
#endif
#ifndef USE_TDM
#  define USE_TDM 0
#endif

union Frag {
    v16h v;
    uint4 q[2];
};
union H8 {
    uint4 q;
    _Float16 h[8];
};

static __device__ inline v8f wmma16(v16h a, v16h b, v8f c) {
    // D = A(16x32 f16) * B(32x16 f16) + C(16x16 f32)
    return __builtin_amdgcn_wmma_f32_16x16x32_f16(
        false, a, false, b, (short)0, c, false, false);
}

#if USE_TDM
// Workgroup-relative LDS byte offset of a __shared__ object.
static __device__ inline unsigned int lds_off(const void* p) {
    typedef __attribute__((address_space(3))) const void* lds_ptr_t;
    return (unsigned int)(unsigned long long)(lds_ptr_t)p;
}

// TDM: load a 128-row x 32-halfword tile (row stride K halves) from global
// into LDS with 16B padding after each 64B row (=> 40-half LDS row stride).
static __device__ inline void tdm_load_tile_128x32(const _Float16* gsrc,
                                                   unsigned int ldsByteOff,
                                                   int K) {
    const unsigned long long ga = (unsigned long long)gsrc;
    v4u g0;
    g0.x = 1u;                                            // count=1, user desc
    g0.y = ldsByteOff;                                    // lds_addr (bytes)
    g0.z = (unsigned int)(ga & 0xFFFFFFFFull);            // global_addr lo
    g0.w = (unsigned int)((ga >> 32) & 0x01FFFFFFull)     // global_addr hi
           | (2u << 30);                                  // type=2 ("image")
    v8i g1;
    g1[0] = (int)((1u << 16)        // data_size = 1 -> 2 bytes/elem
                  | (1u << 20)      // pad_enable
                  | (3u << 22)      // pad_interval: 16 DWORDs (64B) per row
                  | (3u << 25));    // pad_amount: 4 DWORDs (16B)
    g1[1] = (int)(((unsigned)K & 0xFFFFu) << 16);         // tensor_dim0 lo16
    g1[2] = (int)((((unsigned)K >> 16) & 0xFFFFu)         // tensor_dim0 hi16
                  | (128u << 16));                        // tensor_dim1 = 128
    g1[3] = (int)(32u << 16);                             // tile_dim0 = 32
    g1[4] = (int)128;                                     // tile_dim1 = 128
    g1[5] = (int)K;                                       // dim0_stride lo32
    g1[6] = 0;
    g1[7] = 0;
    const v4i z4 = {0, 0, 0, 0};                          // groups 2/3 unused
    const v8i z8 = {0, 0, 0, 0, 0, 0, 0, 0};
    __builtin_amdgcn_tensor_load_to_lds(g0, g1, z4, z4, z8, 0);
}
#endif

// ---------------------------------------------------------------------------
// f32 -> f16 cast (for weights)
// ---------------------------------------------------------------------------
__global__ __launch_bounds__(256) void cast_f32_f16_kernel(
    const float* __restrict__ in, _Float16* __restrict__ out, int n) {
    int i = blockIdx.x * 256 + threadIdx.x;
    if (i < n) out[i] = (_Float16)in[i];
}

// ---------------------------------------------------------------------------
// LayerNorm over last dim (1024), one block (256 thr) per row, f16 output.
// ---------------------------------------------------------------------------
__global__ __launch_bounds__(256) void ln_f16_kernel(
    const float* __restrict__ x, const float* __restrict__ g,
    const float* __restrict__ bta, _Float16* __restrict__ out) {
    const int row = blockIdx.x;
    const int t = threadIdx.x;
    const float* xr = x + (size_t)row * HH;
    float v[4];
    float s = 0.f;
#pragma unroll
    for (int i = 0; i < 4; i++) { v[i] = xr[t + i * 256]; s += v[i]; }
#pragma unroll
    for (int m = 16; m >= 1; m >>= 1) s += __shfl_xor(s, m, 32);
    __shared__ float red[8];
    __shared__ float stat[2];
    const int wid = t >> 5, lane = t & 31;
    if (lane == 0) red[wid] = s;
    __syncthreads();
    if (t == 0) {
        float tot = 0.f;
        for (int i = 0; i < 8; i++) tot += red[i];
        stat[0] = tot * (1.0f / HH);
    }
    __syncthreads();
    const float mean = stat[0];
    float ss = 0.f;
#pragma unroll
    for (int i = 0; i < 4; i++) { float d = v[i] - mean; ss += d * d; }
#pragma unroll
    for (int m = 16; m >= 1; m >>= 1) ss += __shfl_xor(ss, m, 32);
    if (lane == 0) red[wid] = ss;
    __syncthreads();
    if (t == 0) {
        float tot = 0.f;
        for (int i = 0; i < 8; i++) tot += red[i];
        stat[1] = rsqrtf(tot * (1.0f / HH) + 1e-5f);
    }
    __syncthreads();
    const float rs = stat[1];
#pragma unroll
    for (int i = 0; i < 4; i++) {
        int c = t + i * 256;
        out[(size_t)row * HH + c] = (_Float16)((v[i] - mean) * rs * g[c] + bta[c]);
    }
}

// ---------------------------------------------------------------------------
// RoPE + head split: in f32 (B,S,H) -> out f16 (B,NH,S,HD).
// ---------------------------------------------------------------------------
__global__ __launch_bounds__(256) void rope_heads_kernel(
    const float* __restrict__ in, const int* __restrict__ pos,
    _Float16* __restrict__ out, int apply_rope) {
    int idx = blockIdx.x * 256 + threadIdx.x;
    if (idx >= BB * SS * NHH * 32) return;
    const int d = idx & 31;
    const int h = (idx >> 5) & (NHH - 1);
    const int s = (idx >> 9) & (SS - 1);
    const int b = idx >> 20;
    const float* row = in + ((size_t)(b * SS + s)) * HH + h * HDD;
    const float x1 = row[d];
    const float x2 = row[d + 32];
    const size_t ob = ((size_t)((b * NHH + h) * SS + s)) * HDD;
    if (apply_rope) {
        const float p = (float)pos[b * SS + s];
        const float inv = __expf(-(float)d * (9.210340371976184f / 32.0f));
        const float fr = p * inv;
        const float c = cosf(fr), sn = sinf(fr);
        out[ob + d]      = (_Float16)(x1 * c - x2 * sn);
        out[ob + d + 32] = (_Float16)(x2 * c + x1 * sn);
    } else {
        out[ob + d]      = (_Float16)x1;
        out[ob + d + 32] = (_Float16)x2;
    }
}

// ---------------------------------------------------------------------------
// WMMA GEMM: out(M,N) = A(M,K) * W(N,K)^T + bias, fused epilogue.
// EPI: 0 = f32 store, 1 = f16 store, 2 = GELU -> f16, 3 = +res -> f32.
// Block 256 thr (8 waves, 2x4), tile 128x128, BK=32.
// TDM path: double-buffered LDS tiles filled by tensor_load_to_lds.
// ---------------------------------------------------------------------------
template <int EPI>
__global__ __launch_bounds__(256) void gemm_wmma_kernel(
    const _Float16* __restrict__ A, const _Float16* __restrict__ W,
    const float* __restrict__ bias, const float* __restrict__ res,
    float* __restrict__ outF, _Float16* __restrict__ outH,
    int M, int N, int K) {
    __shared__ _Float16 As[2][128 * 40];
    __shared__ _Float16 Bs[2][128 * 40];

    const int tid = threadIdx.x;
    const int bm = blockIdx.y * 128;
    const int bn = blockIdx.x * 128;
    const int wid = tid >> 5, lane = tid & 31;
    const int wm = wid >> 2, wn = wid & 3;          // 2 x 4 wave grid
    const int l15 = lane & 15, lh = (lane >> 4) & 1;

    const v8f zf = {0.f, 0.f, 0.f, 0.f, 0.f, 0.f, 0.f, 0.f};
    v8f acc[4][2];
#pragma unroll
    for (int i = 0; i < 4; i++)
#pragma unroll
        for (int j = 0; j < 2; j++) acc[i][j] = zf;

    auto compute_tile = [&](const _Float16* Acur, const _Float16* Bcur) {
        Frag fa[4];
#pragma unroll
        for (int mt = 0; mt < 4; mt++) {
            const _Float16* p = Acur + (wm * 64 + mt * 16 + l15) * 40 + lh * 8;
            fa[mt].q[0] = *(const uint4*)(p);
            fa[mt].q[1] = *(const uint4*)(p + 16);
        }
        Frag fb[2];
#pragma unroll
        for (int nt = 0; nt < 2; nt++) {
            const _Float16* p = Bcur + (wn * 32 + nt * 16 + l15) * 40 + lh * 8;
            fb[nt].q[0] = *(const uint4*)(p);
            fb[nt].q[1] = *(const uint4*)(p + 16);
        }
#pragma unroll
        for (int mt = 0; mt < 4; mt++)
#pragma unroll
            for (int nt = 0; nt < 2; nt++)
                acc[mt][nt] = wmma16(fa[mt].v, fb[nt].v, acc[mt][nt]);
    };

#if USE_TDM
    const int nk = K >> 5;
    const _Float16* gAbase = A + (size_t)bm * K;
    const _Float16* gBbase = W + (size_t)bn * K;
    const unsigned aOff[2] = {lds_off(&As[0][0]), lds_off(&As[1][0])};
    const unsigned bOff[2] = {lds_off(&Bs[0][0]), lds_off(&Bs[1][0])};

    if (wid == 0) {
        tdm_load_tile_128x32(gAbase, aOff[0], K);
        tdm_load_tile_128x32(gBbase, bOff[0], K);
    }
    for (int kt = 0; kt < nk; kt++) {
        const int cur = kt & 1;
        if (wid == 0) {
            if (kt + 1 < nk) {
                tdm_load_tile_128x32(gAbase + (size_t)(kt + 1) * 32,
                                     aOff[cur ^ 1], K);
                tdm_load_tile_128x32(gBbase + (size_t)(kt + 1) * 32,
                                     bOff[cur ^ 1], K);
                __builtin_amdgcn_s_wait_tensorcnt(2);  // tile kt's pair done
            } else {
                __builtin_amdgcn_s_wait_tensorcnt(0);
            }
        }
        __syncthreads();                     // data visible to all waves
        compute_tile(&As[cur][0], &Bs[cur][0]);
        __syncthreads();                     // done reading before overwrite
    }
#else
    const int lr = tid >> 1, seg = tid & 1;          // 128 rows x 2 segments
    const _Float16* gA = A + (size_t)(bm + lr) * K + seg * 16;
    const _Float16* gB = W + (size_t)(bn + lr) * K + seg * 16;
    _Float16* sA = &As[0][0] + lr * 40 + seg * 16;
    _Float16* sB = &Bs[0][0] + lr * 40 + seg * 16;

    for (int k0 = 0; k0 < K; k0 += 32) {
        const uint4 a0 = *(const uint4*)(gA + k0);
        const uint4 a1 = *(const uint4*)(gA + k0 + 8);
        const uint4 b0 = *(const uint4*)(gB + k0);
        const uint4 b1 = *(const uint4*)(gB + k0 + 8);
        if (k0 + 32 < K) {
            __builtin_prefetch(gA + k0 + 32, 0, 0);
            __builtin_prefetch(gB + k0 + 32, 0, 0);
        }
        __syncthreads();
        *(uint4*)(sA) = a0; *(uint4*)(sA + 8) = a1;
        *(uint4*)(sB) = b0; *(uint4*)(sB + 8) = b1;
        __syncthreads();
        compute_tile(&As[0][0], &Bs[0][0]);
    }
#endif

#pragma unroll
    for (int mt = 0; mt < 4; mt++) {
#pragma unroll
        for (int nt = 0; nt < 2; nt++) {
#pragma unroll
            for (int r = 0; r < 8; r++) {
                const int row = bm + wm * 64 + mt * 16 + r + lh * 8;
                const int col = bn + wn * 32 + nt * 16 + l15;
                float v = acc[mt][nt][r] + bias[col];
                const size_t idx = (size_t)row * N + col;
                if (EPI == 0) {
                    outF[idx] = v;
                } else if (EPI == 1) {
                    outH[idx] = (_Float16)v;
                } else if (EPI == 2) {
                    v = 0.5f * v * (1.0f + erff(v * 0.70710678118654752f));
                    outH[idx] = (_Float16)v;
                } else {
                    outF[idx] = v + res[idx];
                }
            }
        }
    }
}

// ---------------------------------------------------------------------------
// Flash attention: Q,K,V f16 (B,NH,S,HD) -> attn f16 (B,S,H).
// Block 128 thr (4 waves); 64-query tile, stream 64-key tiles; online softmax.
// ---------------------------------------------------------------------------
__global__ __launch_bounds__(128) void flash_attn_kernel(
    const _Float16* __restrict__ Q, const _Float16* __restrict__ K,
    const _Float16* __restrict__ V, const unsigned char* __restrict__ mask,
    _Float16* __restrict__ outA) {
    __shared__ _Float16 Qs[64 * 72];
    __shared__ _Float16 Ks[64 * 72];
    __shared__ _Float16 Vt[64 * 72];
    __shared__ _Float16 Ps[64 * 72];

    const int tid = threadIdx.x;
    const int bh = blockIdx.y;            // 0..31
    const int b = bh >> 4, h = bh & 15;
    const int qb = blockIdx.x;            // 0..31
    const int wid = tid >> 5, lane = tid & 31;
    const int l15 = lane & 15, lh = (lane >> 4) & 1;
    const int lr = tid >> 1, seg = tid & 1;

    // load Q tile (64 x 64)
    {
        const _Float16* gq = Q + ((size_t)bh * SS + qb * 64 + lr) * HDD + seg * 32;
        _Float16* sq = Qs + lr * 72 + seg * 32;
#pragma unroll
        for (int j = 0; j < 4; j++)
            *(uint4*)(sq + j * 8) = *(const uint4*)(gq + j * 8);
    }

    const v8f zf = {0.f, 0.f, 0.f, 0.f, 0.f, 0.f, 0.f, 0.f};
    v8f o[4];
#pragma unroll
    for (int i = 0; i < 4; i++) o[i] = zf;
    float mst[8], lst[8];
#pragma unroll
    for (int r = 0; r < 8; r++) { mst[r] = -INFINITY; lst[r] = 0.f; }

    for (int kb = 0; kb < SS / 64; kb++) {
        __syncthreads();
        // K tile row-major; V tile transposed (d-major) for B-frag loads
        {
            const _Float16* gk = K + ((size_t)bh * SS + kb * 64 + lr) * HDD + seg * 32;
            _Float16* sk = Ks + lr * 72 + seg * 32;
#pragma unroll
            for (int j = 0; j < 4; j++)
                *(uint4*)(sk + j * 8) = *(const uint4*)(gk + j * 8);
            const _Float16* gv = V + ((size_t)bh * SS + kb * 64 + lr) * HDD + seg * 32;
#pragma unroll
            for (int j = 0; j < 4; j++) {
                H8 t8; t8.q = *(const uint4*)(gv + j * 8);
#pragma unroll
                for (int i = 0; i < 8; i++)
                    Vt[(seg * 32 + j * 8 + i) * 72 + lr] = t8.h[i];
            }
        }
        __syncthreads();

        // scores S = Q * K^T
        v8f sc[4];
#pragma unroll
        for (int i = 0; i < 4; i++) sc[i] = zf;
#pragma unroll
        for (int k0 = 0; k0 < 64; k0 += 32) {
            Frag fa;
            {
                const _Float16* p = Qs + (wid * 16 + l15) * 72 + k0 + lh * 8;
                fa.q[0] = *(const uint4*)(p);
                fa.q[1] = *(const uint4*)(p + 16);
            }
#pragma unroll
            for (int nt = 0; nt < 4; nt++) {
                Frag fb;
                const _Float16* p = Ks + (nt * 16 + l15) * 72 + k0 + lh * 8;
                fb.q[0] = *(const uint4*)(p);
                fb.q[1] = *(const uint4*)(p + 16);
                sc[nt] = wmma16(fa.v, fb.v, sc[nt]);
            }
        }

        // online softmax (per-row reductions over each 16-lane half-wave)
#pragma unroll
        for (int r = 0; r < 8; r++) {
            float rm = -INFINITY;
#pragma unroll
            for (int nt = 0; nt < 4; nt++) {
                float sv = sc[nt][r] * 0.125f;
                const int key = kb * 64 + nt * 16 + l15;
                if (mask[b * SS + key] == 0) sv = -INFINITY;
                sc[nt][r] = sv;
                rm = fmaxf(rm, sv);
            }
#pragma unroll
            for (int m = 8; m >= 1; m >>= 1) rm = fmaxf(rm, __shfl_xor(rm, m, 32));
            const float mnew = fmaxf(mst[r], rm);
            const float al = (mnew == -INFINITY) ? 1.0f : __expf(mst[r] - mnew);
            float ps = 0.f;
            const int prow = wid * 16 + r + lh * 8;
#pragma unroll
            for (int nt = 0; nt < 4; nt++) {
                const float p = (mnew == -INFINITY) ? 0.f : __expf(sc[nt][r] - mnew);
                ps += p;
                Ps[prow * 72 + nt * 16 + l15] = (_Float16)p;
            }
#pragma unroll
            for (int m = 8; m >= 1; m >>= 1) ps += __shfl_xor(ps, m, 32);
            lst[r] = lst[r] * al + ps;
            mst[r] = mnew;
#pragma unroll
            for (int nt = 0; nt < 4; nt++) o[nt][r] *= al;
        }
        __syncthreads();

        // O += P * V
#pragma unroll
        for (int k0 = 0; k0 < 64; k0 += 32) {
            Frag fa;
            {
                const _Float16* p = Ps + (wid * 16 + l15) * 72 + k0 + lh * 8;
                fa.q[0] = *(const uint4*)(p);
                fa.q[1] = *(const uint4*)(p + 16);
            }
#pragma unroll
            for (int nt = 0; nt < 4; nt++) {
                Frag fb;
                const _Float16* p = Vt + (nt * 16 + l15) * 72 + k0 + lh * 8;
                fb.q[0] = *(const uint4*)(p);
                fb.q[1] = *(const uint4*)(p + 16);
                o[nt] = wmma16(fa.v, fb.v, o[nt]);
            }
        }
    }

    // normalize + store to (B,S,H) f16
#pragma unroll
    for (int r = 0; r < 8; r++) {
        const float inv = (lst[r] > 0.f) ? (1.0f / lst[r]) : 0.f;
        const int srow = qb * 64 + wid * 16 + r + lh * 8;
#pragma unroll
        for (int nt = 0; nt < 4; nt++) {
            const int col = h * HDD + nt * 16 + l15;
            outA[((size_t)(b * SS + srow)) * HH + col] = (_Float16)(o[nt][r] * inv);
        }
    }
}

// ---------------------------------------------------------------------------
// Host-side orchestration
// ---------------------------------------------------------------------------
static constexpr size_t MB = 1ull << 20;
static constexpr size_t OFF_WQ   = 0;          // 2 MB  (H*H f16)
static constexpr size_t OFF_WK   = 2 * MB;
static constexpr size_t OFF_WV   = 4 * MB;
static constexpr size_t OFF_WO   = 6 * MB;
static constexpr size_t OFF_W1   = 8 * MB;     // 8 MB  (FF*H f16)
static constexpr size_t OFF_W2   = 16 * MB;    // 8 MB
static constexpr size_t OFF_HBF  = 24 * MB;    // 8 MB  (B*S*H f16)
static constexpr size_t OFF_TMPF = 32 * MB;    // 16 MB (B*S*H f32)
static constexpr size_t OFF_QBH  = 48 * MB;    // 8 MB
static constexpr size_t OFF_KBH  = 56 * MB;    // 8 MB
static constexpr size_t OFF_VBH  = 64 * MB;    // 8 MB
static constexpr size_t OFF_ATTN = 72 * MB;    // 8 MB
static constexpr size_t OFF_H2   = 80 * MB;    // 8 MB
static constexpr size_t OFF_F1   = 88 * MB;    // 32 MB (B*S*FF f16)

extern "C" void kernel_launch(void* const* d_in, const int* in_sizes, int n_in,
                              void* d_out, int out_size, void* d_ws, size_t ws_size,
                              hipStream_t stream) {
    (void)in_sizes; (void)n_in; (void)out_size; (void)ws_size;
    const float* x     = (const float*)d_in[0];
    const float* Wq    = (const float*)d_in[1];
    const float* bq    = (const float*)d_in[2];
    const float* Wk    = (const float*)d_in[3];
    const float* bk    = (const float*)d_in[4];
    const float* Wv    = (const float*)d_in[5];
    const float* bv    = (const float*)d_in[6];
    const float* Wo    = (const float*)d_in[7];
    const float* bo    = (const float*)d_in[8];
    const float* W1    = (const float*)d_in[9];
    const float* b1    = (const float*)d_in[10];
    const float* W2    = (const float*)d_in[11];
    const float* b2    = (const float*)d_in[12];
    const float* g1    = (const float*)d_in[13];
    const float* be1   = (const float*)d_in[14];
    const float* g2    = (const float*)d_in[15];
    const float* be2   = (const float*)d_in[16];
    const unsigned char* mask = (const unsigned char*)d_in[17];
    const int* pos     = (const int*)d_in[18];

    char* ws = (char*)d_ws;
    _Float16* wq_h  = (_Float16*)(ws + OFF_WQ);
    _Float16* wk_h  = (_Float16*)(ws + OFF_WK);
    _Float16* wv_h  = (_Float16*)(ws + OFF_WV);
    _Float16* wo_h  = (_Float16*)(ws + OFF_WO);
    _Float16* w1_h  = (_Float16*)(ws + OFF_W1);
    _Float16* w2_h  = (_Float16*)(ws + OFF_W2);
    _Float16* hbf   = (_Float16*)(ws + OFF_HBF);
    float*    tmpf  = (float*)(ws + OFF_TMPF);
    _Float16* qbh   = (_Float16*)(ws + OFF_QBH);
    _Float16* kbh   = (_Float16*)(ws + OFF_KBH);
    _Float16* vbh   = (_Float16*)(ws + OFF_VBH);
    _Float16* attnb = (_Float16*)(ws + OFF_ATTN);
    _Float16* h2bf  = (_Float16*)(ws + OFF_H2);
    _Float16* f1bf  = (_Float16*)(ws + OFF_F1);
    float*    outF  = (float*)d_out;

    const int nHH = HH * HH;          // 1M
    const int nFH = FFF * HH;         // 4M
    cast_f32_f16_kernel<<<nHH / 256, 256, 0, stream>>>(Wq, wq_h, nHH);
    cast_f32_f16_kernel<<<nHH / 256, 256, 0, stream>>>(Wk, wk_h, nHH);
    cast_f32_f16_kernel<<<nHH / 256, 256, 0, stream>>>(Wv, wv_h, nHH);
    cast_f32_f16_kernel<<<nHH / 256, 256, 0, stream>>>(Wo, wo_h, nHH);
    cast_f32_f16_kernel<<<nFH / 256, 256, 0, stream>>>(W1, w1_h, nFH);
    cast_f32_f16_kernel<<<nFH / 256, 256, 0, stream>>>(W2, w2_h, nFH);

    const int Mrows = BB * SS;  // 4096
    ln_f16_kernel<<<Mrows, 256, 0, stream>>>(x, g1, be1, hbf);

    const dim3 gHH(HH / 128, Mrows / 128);   // (8, 32)
    const dim3 gFF(FFF / 128, Mrows / 128);  // (32, 32)
    const int ropeBlocks = (BB * SS * NHH * 32) / 256;  // 8192

    // Q
    gemm_wmma_kernel<0><<<gHH, 256, 0, stream>>>(hbf, wq_h, bq, nullptr, tmpf, nullptr, Mrows, HH, HH);
    rope_heads_kernel<<<ropeBlocks, 256, 0, stream>>>(tmpf, pos, qbh, 1);
    // K
    gemm_wmma_kernel<0><<<gHH, 256, 0, stream>>>(hbf, wk_h, bk, nullptr, tmpf, nullptr, Mrows, HH, HH);
    rope_heads_kernel<<<ropeBlocks, 256, 0, stream>>>(tmpf, pos, kbh, 1);
    // V
    gemm_wmma_kernel<0><<<gHH, 256, 0, stream>>>(hbf, wv_h, bv, nullptr, tmpf, nullptr, Mrows, HH, HH);
    rope_heads_kernel<<<ropeBlocks, 256, 0, stream>>>(tmpf, pos, vbh, 0);

    // attention
    flash_attn_kernel<<<dim3(SS / 64, BB * NHH), 128, 0, stream>>>(qbh, kbh, vbh, mask, attnb);

    // x1 = x + attn @ Wo^T + bo  -> d_out
    gemm_wmma_kernel<3><<<gHH, 256, 0, stream>>>(attnb, wo_h, bo, x, outF, nullptr, Mrows, HH, HH);

    // h2 = LN(x1)
    ln_f16_kernel<<<Mrows, 256, 0, stream>>>(outF, g2, be2, h2bf);

    // f1 = gelu(h2 @ W1^T + b1)  (f16)
    gemm_wmma_kernel<2><<<gFF, 256, 0, stream>>>(h2bf, w1_h, b1, nullptr, nullptr, f1bf, Mrows, FFF, HH);

    // out = x1 + f1 @ W2^T + b2
    gemm_wmma_kernel<3><<<gHH, 256, 0, stream>>>(f1bf, w2_h, b2, outF, outF, nullptr, Mrows, HH, FFF);
}
